// MultiHead_Attention_1752346657142
// MI455X (gfx1250) — matrix operand order repaired and checked by reading, not compiled, not run on hardware
//
#include <hip/hip_runtime.h>
#include <stdint.h>

// ---------------- types ----------------
typedef __bf16 bf16;
typedef __attribute__((ext_vector_type(16))) __bf16 v16bf;
typedef __attribute__((ext_vector_type(8)))  __bf16 v8bf;
typedef __attribute__((ext_vector_type(8)))  float  v8f;

#define B_     2
#define LQ_    1024
#define LK_    1024
#define DH_    64
#define H_     16
#define DM_    1024
#define USEL_  34
#define USAMP_ 7097
#define NTILES_ ((USAMP_ + 15) / 16)   // 444

// ---------------- WMMA fragment loaders (contiguous k order, k = 16*(lane/16) + s) ----------------
// A (16x32 bf16): lane M holds K=0..15 ; lane M+16 holds K=16..31.
__device__ __forceinline__ v16bf load_a_frag(const bf16* rowbase, int lane) {
  const v8bf lo = *(const v8bf*)(rowbase + ((lane >> 4) << 4));
  const v8bf hi = *(const v8bf*)(rowbase + 8 + ((lane >> 4) << 4));
  v16bf a;
#pragma unroll
  for (int i = 0; i < 8; ++i) { a[i] = lo[i]; a[i + 8] = hi[i]; }
  return a;
}
// B (32x16 bf16): lane N (0-15) holds K=0..15 of column N; lane N+16 holds K=16..31.
__device__ __forceinline__ v16bf load_b_frag(const bf16* colbase, int lane) {
  return *(const v16bf*)(colbase + ((lane >> 4) << 4));
}

__device__ __forceinline__ v8f wmma_bf16(v16bf a, v16bf b, v8f c) {
  return __builtin_amdgcn_wmma_f32_16x16x32_bf16(false, a, false, b, (short)0, c, false, false);
}

// ---------------- small utility kernels ----------------
__global__ void cvt_bf16_kernel(const float* __restrict__ s, bf16* __restrict__ d, int n) {
  for (int i = blockIdx.x * blockDim.x + threadIdx.x; i < n; i += gridDim.x * blockDim.x)
    d[i] = (bf16)s[i];
}

// Wt[n][k] = bf16(W[k][n])  (so projection B-fragments are contiguous per lane)
__global__ void __launch_bounds__(256) transpose_w_kernel(const float* __restrict__ W,
                                                          bf16* __restrict__ Wt) {
  __shared__ bf16 tile[32][33];
  const int bx = blockIdx.x * 32, by = blockIdx.y * 32;
  const int tx = threadIdx.x & 31, ty = threadIdx.x >> 5;
  for (int i = ty; i < 32; i += 8)
    tile[i][tx] = (bf16)W[(size_t)(by + i) * DM_ + bx + tx];
  __syncthreads();
  for (int i = ty; i < 32; i += 8)
    Wt[(size_t)(bx + i) * DM_ + by + tx] = tile[tx][i];
}

__global__ void vmean_kernel(const float* __restrict__ v, float* __restrict__ vmean) {
  const int b = blockIdx.x, d = threadIdx.x;  // 64 threads
  float s = 0.f;
  for (int k = 0; k < LK_; ++k) s += v[((size_t)b * LK_ + k) * DH_ + d];
  vmean[b * DH_ + d] = s * (1.0f / (float)LK_);
}

// attn[b][q][h*64+d] = mean(v)[b][d] for every head (v shared across heads)
__global__ void fill_mean_kernel(const float* __restrict__ vmean, bf16* __restrict__ attn) {
  const unsigned total = B_ * LQ_ * DM_;
  for (unsigned i = blockIdx.x * blockDim.x + threadIdx.x; i < total;
       i += gridDim.x * blockDim.x) {
    const unsigned b = i >> 20;         // LQ_*DM_ = 2^20
    const unsigned c = i & (DM_ - 1);
    attn[i] = (bf16)vmean[b * DH_ + (c & (DH_ - 1))];
  }
}

__device__ __forceinline__ uint32_t mix32(uint32_t x) {
  x ^= x >> 16; x *= 0x7feb352du; x ^= x >> 15; x *= 0x846ca68bu; x ^= x >> 16;
  return x;
}

// ---------------- heavy kernel: sparsity measure M = max - mean of sampled scores --------
// grid (LQ/32, B, H), 8 waves. Each wave owns a 32-row query tile and iterates over its
// share of 444 sampled-key column tiles; each gathered B fragment pair feeds 4 WMMAs
// (2 row tiles x K=64). Software-pipelined so the gather latency hides under the
// matrix pipe; accumulation (v_max_num/v_pk_add) dual-issues in the WMMA shadow.
__global__ void __launch_bounds__(256) mscore_kernel(const bf16* __restrict__ qb,
                                                     const bf16* __restrict__ kb,
                                                     float* __restrict__ Mout) {
  const int h = blockIdx.z, b = blockIdx.y, m0 = blockIdx.x * 32;
  const int wave = threadIdx.x >> 5, lane = threadIdx.x & 31;
  const int coll = lane & 15;

  const bf16* q0 = qb + ((size_t)(b * LQ_ + m0 + coll)) * DH_;
  const bf16* q1 = qb + ((size_t)(b * LQ_ + m0 + 16 + coll)) * DH_;
  const v16bf a00 = load_a_frag(q0, lane),      a01 = load_a_frag(q0 + 32, lane);
  const v16bf a10 = load_a_frag(q1, lane),      a11 = load_a_frag(q1 + 32, lane);

  float rmax[2][8], rsum[2][8];
#pragma unroll
  for (int j = 0; j < 2; ++j)
#pragma unroll
    for (int r = 0; r < 8; ++r) { rmax[j][r] = -1e30f; rsum[j][r] = 0.f; }

  const uint32_t seed = (uint32_t)h * 0x85EBCA6Bu ^ (uint32_t)b * 0xC2B2AE35u;
  const bf16* kbb = kb + (size_t)b * LK_ * DH_;

  // hashed gather pointer for column tile t (this lane's column = t*16 + coll)
  auto rowptr = [&](int t) -> const bf16* {
    const uint32_t u = (uint32_t)(t * 16 + coll);
    return kbb + (size_t)(mix32(u * 0x9E3779B9u ^ seed) & (LK_ - 1)) * DH_;
  };

  // -------- full tiles (all 16 columns valid), 2-stage pipeline --------
  {
    const bf16* kr = rowptr(wave);
    v16bf b0 = load_b_frag(kr, lane);
    v16bf b1 = load_b_frag(kr + 32, lane);
    for (int t = wave; t < NTILES_ - 1; t += 8) {
      // branch-free next-tile index (wraps to this wave's first tile on last iter)
      const int tn = (t + 8 < NTILES_ - 1) ? (t + 8) : wave;
      const bf16* krn = rowptr(tn);
      const v16bf nb0 = load_b_frag(krn, lane);      // issued before current WMMAs;
      const v16bf nb1 = load_b_frag(krn + 32, lane); // wait lands next iteration
      v8f c0 = {}, c1 = {};
      c0 = wmma_bf16(a00, b0, c0);
      c1 = wmma_bf16(a10, b0, c1);
      c0 = wmma_bf16(a01, b1, c0);
      c1 = wmma_bf16(a11, b1, c1);
#pragma unroll
      for (int r = 0; r < 8; ++r) {
        rmax[0][r] = fmaxf(rmax[0][r], c0[r]); rsum[0][r] += c0[r];
        rmax[1][r] = fmaxf(rmax[1][r], c1[r]); rsum[1][r] += c1[r];
      }
      b0 = nb0; b1 = nb1;
    }
  }

  // -------- tail tile (columns >= USAMP_ masked) --------
  if (wave == (NTILES_ - 1) % 8) {
    const bf16* kr = rowptr(NTILES_ - 1);
    const v16bf b0 = load_b_frag(kr, lane);
    const v16bf b1 = load_b_frag(kr + 32, lane);
    v8f c0 = {}, c1 = {};
    c0 = wmma_bf16(a00, b0, c0);
    c1 = wmma_bf16(a10, b0, c1);
    c0 = wmma_bf16(a01, b1, c0);
    c1 = wmma_bf16(a11, b1, c1);
    const bool valid = ((NTILES_ - 1) * 16 + coll < USAMP_);
#pragma unroll
    for (int r = 0; r < 8; ++r) {
      if (valid) {
        rmax[0][r] = fmaxf(rmax[0][r], c0[r]); rsum[0][r] += c0[r];
        rmax[1][r] = fmaxf(rmax[1][r], c1[r]); rsum[1][r] += c1[r];
      }
    }
  }

  // reduce across the 16 lanes holding the same row (stay within each half-wave)
#pragma unroll
  for (int m = 1; m < 16; m <<= 1) {
#pragma unroll
    for (int j = 0; j < 2; ++j)
#pragma unroll
      for (int r = 0; r < 8; ++r) {
        rmax[j][r] = fmaxf(rmax[j][r], __shfl_xor(rmax[j][r], m, 32));
        rsum[j][r] += __shfl_xor(rsum[j][r], m, 32);
      }
  }

  __shared__ float smax[8][32], ssum[8][32];
  if ((lane & 15) == 0) {
    const int rbase = (lane >> 4) * 8;  // lanes 0-15: rows r ; lanes 16-31: rows r+8
#pragma unroll
    for (int r = 0; r < 8; ++r) {
      smax[wave][rbase + r]      = rmax[0][r]; ssum[wave][rbase + r]      = rsum[0][r];
      smax[wave][16 + rbase + r] = rmax[1][r]; ssum[wave][16 + rbase + r] = rsum[1][r];
    }
  }
  __syncthreads();
  if (threadIdx.x < 32) {
    float mx = -1e30f, sm = 0.f;
    for (int w = 0; w < 8; ++w) { mx = fmaxf(mx, smax[w][threadIdx.x]); sm += ssum[w][threadIdx.x]; }
    Mout[((size_t)(h * B_ + b)) * LQ_ + m0 + threadIdx.x] = mx - sm * (1.0f / (float)USAMP_);
  }
}

// ---------------- top-k (iterative block argmax) ----------------
__global__ void __launch_bounds__(256) topk_kernel(const float* __restrict__ M,
                                                   int* __restrict__ topidx) {
  const int hb = blockIdx.x;
  __shared__ float sm[LQ_];
  __shared__ float rv[256];
  __shared__ int ri[256];
  const int t = threadIdx.x;
  for (int i = t; i < LQ_; i += 256) sm[i] = M[(size_t)hb * LQ_ + i];
  __syncthreads();
  for (int it = 0; it < USEL_; ++it) {
    float bv = -1e30f; int bi = 0;
    for (int i = t; i < LQ_; i += 256) { const float x = sm[i]; if (x > bv) { bv = x; bi = i; } }
    rv[t] = bv; ri[t] = bi; __syncthreads();
    for (int s = 128; s > 0; s >>= 1) {
      if (t < s && rv[t + s] > rv[t]) { rv[t] = rv[t + s]; ri[t] = ri[t + s]; }
      __syncthreads();
    }
    if (t == 0) { topidx[hb * 64 + it] = ri[0]; sm[ri[0]] = -1e30f; }
    __syncthreads();
  }
}

// ---------------- attention on the 34 selected queries (tiny; VALU fp32) ----------------
__global__ void __launch_bounds__(256) sel_attn_kernel(const float* __restrict__ q,
                                                       const bf16* __restrict__ kb,
                                                       const bf16* __restrict__ vb,
                                                       const int* __restrict__ topidx,
                                                       bf16* __restrict__ attn) {
  const int hb = blockIdx.x, h = hb >> 1, b = hb & 1;
  __shared__ float qrow[DH_];
  __shared__ float p[LK_];
  __shared__ float red[256];
  __shared__ float red2[256];
  const int t = threadIdx.x;
  for (int it = 0; it < USEL_; ++it) {
    const int qi = topidx[hb * 64 + it];
    if (t < DH_) qrow[t] = q[((size_t)b * LQ_ + qi) * DH_ + t];
    __syncthreads();
    for (int k = t; k < LK_; k += 256) {
      const bf16* kr = kb + ((size_t)b * LK_ + k) * DH_;
      float d = 0.f;
#pragma unroll 8
      for (int c = 0; c < DH_; ++c) d += qrow[c] * (float)kr[c];
      p[k] = d * 0.03125f;  // 1/sqrt(LK)=1/32
    }
    __syncthreads();
    float lm = -1e30f;
    for (int k = t; k < LK_; k += 256) lm = fmaxf(lm, p[k]);
    red[t] = lm; __syncthreads();
    for (int s = 128; s > 0; s >>= 1) { if (t < s) red[t] = fmaxf(red[t], red[t + s]); __syncthreads(); }
    const float gmax = red[0]; __syncthreads();
    float ls = 0.f;
    for (int k = t; k < LK_; k += 256) { const float e = __expf(p[k] - gmax); p[k] = e; ls += e; }
    red[t] = ls; __syncthreads();
    for (int s = 128; s > 0; s >>= 1) { if (t < s) red[t] += red[t + s]; __syncthreads(); }
    const float inv = 1.0f / red[0]; __syncthreads();
    // out[d] = sum_k p[k]*v[k][d] ; 4 k-chunks x 64 dims
    const int d = t & 63, ch = t >> 6;
    float acc = 0.f;
    for (int k = ch * 256; k < ch * 256 + 256; ++k)
      acc += p[k] * (float)vb[((size_t)b * LK_ + k) * DH_ + d];
    red2[t] = acc; __syncthreads();
    if (t < DH_) {
      const float o = (red2[t] + red2[t + 64] + red2[t + 128] + red2[t + 192]) * inv;
      attn[((size_t)(b * LQ_ + qi)) * DM_ + h * DH_ + t] = (bf16)o;
    }
    __syncthreads();
  }
}

// ---------------- projection GEMM: out(2048x1024) = attn * W + bias (bf16 WMMA) -------
// grid (DM/128, 2048/64), 8 waves arranged 2x4; each wave computes a 32x32 tile:
// 4 accumulators, 4 WMMA per K-step of 32 (load:wmma = 1:1). Unroll 2 so the compiler
// interleaves the next step's fragment loads under the current step's WMMAs.
__global__ void __launch_bounds__(256) proj_kernel(const bf16* __restrict__ A,
                                                   const bf16* __restrict__ Wt,
                                                   const float* __restrict__ bias,
                                                   float* __restrict__ out) {
  const int n0 = blockIdx.x * 128, m0 = blockIdx.y * 64;
  const int wave = threadIdx.x >> 5, lane = threadIdx.x & 31;
  const int mw = m0 + (wave >> 2) * 32;
  const int nw = n0 + (wave & 3) * 32;
  const bf16* ar0 = A + (size_t)(mw + (lane & 15)) * DM_;
  const bf16* ar1 = A + (size_t)(mw + 16 + (lane & 15)) * DM_;
  const bf16* bc0 = Wt + (size_t)(nw + (lane & 15)) * DM_;
  const bf16* bc1 = Wt + (size_t)(nw + 16 + (lane & 15)) * DM_;
  v8f acc00 = {}, acc01 = {}, acc10 = {}, acc11 = {};
#pragma unroll 2
  for (int k0 = 0; k0 < DM_; k0 += 32) {
    const v16bf a0 = load_a_frag(ar0 + k0, lane);
    const v16bf a1 = load_a_frag(ar1 + k0, lane);
    const v16bf b0 = load_b_frag(bc0 + k0, lane);
    const v16bf b1 = load_b_frag(bc1 + k0, lane);
    acc00 = wmma_bf16(a0, b0, acc00);
    acc01 = wmma_bf16(a0, b1, acc01);
    acc10 = wmma_bf16(a1, b0, acc10);
    acc11 = wmma_bf16(a1, b1, acc11);
  }
  const int rbase = (lane >> 4) * 8;
  const int c0 = nw + (lane & 15), c1 = c0 + 16;
  const float bias0 = bias[c0], bias1 = bias[c1];
#pragma unroll
  for (int r = 0; r < 8; ++r) {
    const int row0 = mw + r + rbase;
    const int row1 = row0 + 16;
    out[(size_t)row0 * DM_ + c0] = acc00[r] + bias0;
    out[(size_t)row0 * DM_ + c1] = acc01[r] + bias1;
    out[(size_t)row1 * DM_ + c0] = acc10[r] + bias0;
    out[(size_t)row1 * DM_ + c1] = acc11[r] + bias1;
  }
}

// ---------------- host ----------------
extern "C" void kernel_launch(void* const* d_in, const int* in_sizes, int n_in,
                              void* d_out, int out_size, void* d_ws, size_t ws_size,
                              hipStream_t stream) {
  const float* q    = (const float*)d_in[0];
  const float* k    = (const float*)d_in[1];
  const float* v    = (const float*)d_in[2];
  const float* W    = (const float*)d_in[3];
  const float* bias = (const float*)d_in[4];
  float* out = (float*)d_out;

  char* ws = (char*)d_ws;
  size_t off = 0;
  auto carve = [&](size_t bytes) -> void* {
    void* p = ws + off;
    off = (off + bytes + 255) & ~(size_t)255;
    return p;
  };
  bf16* qb    = (bf16*)carve((size_t)B_ * LQ_ * DH_ * sizeof(bf16));
  bf16* kb    = (bf16*)carve((size_t)B_ * LK_ * DH_ * sizeof(bf16));
  bf16* vb    = (bf16*)carve((size_t)B_ * LK_ * DH_ * sizeof(bf16));
  bf16* Wt    = (bf16*)carve((size_t)DM_ * DM_ * sizeof(bf16));
  bf16* attn  = (bf16*)carve((size_t)B_ * LQ_ * DM_ * sizeof(bf16));
  float* Mbuf = (float*)carve((size_t)H_ * B_ * LQ_ * sizeof(float));
  int* topidx = (int*)carve((size_t)H_ * B_ * 64 * sizeof(int));
  float* vmean = (float*)carve((size_t)B_ * DH_ * sizeof(float));

  cvt_bf16_kernel<<<256, 256, 0, stream>>>(q, qb, B_ * LQ_ * DH_);
  cvt_bf16_kernel<<<256, 256, 0, stream>>>(k, kb, B_ * LK_ * DH_);
  cvt_bf16_kernel<<<256, 256, 0, stream>>>(v, vb, B_ * LK_ * DH_);
  transpose_w_kernel<<<dim3(DM_ / 32, DM_ / 32), 256, 0, stream>>>(W, Wt);
  vmean_kernel<<<B_, DH_, 0, stream>>>(v, vmean);
  fill_mean_kernel<<<2048, 256, 0, stream>>>(vmean, attn);
  mscore_kernel<<<dim3(LQ_ / 32, B_, H_), 256, 0, stream>>>(qb, kb, Mbuf);
  topk_kernel<<<H_ * B_, 256, 0, stream>>>(Mbuf, topidx);
  sel_attn_kernel<<<H_ * B_, 256, 0, stream>>>(q, kb, vb, topidx, attn);
  proj_kernel<<<dim3(DM_ / 128, (B_ * LQ_) / 64), 256, 0, stream>>>(attn, Wt, bias, out);
}
